// MSA_87497073754961
// MI455X (gfx1250) — matrix operand order repaired and checked by reading, not compiled, not run on hardware
//
#include <hip/hip_runtime.h>
#include <hip/hip_bf16.h>
#include <math.h>

// ---------------------------------------------------------------------------
// MSA forward for B=4, S=2048, D=256, H=8, DH=32 on gfx1250 (MI455X).
// bf16 WMMA (v_wmma_f32_16x16x32_bf16), f32 accumulation, flash-attention
// streaming (the S x S score matrix never touches HBM), K tiles staged to
// LDS by the Tensor Data Mover with double buffering, V read transposed
// straight from L2.
// ---------------------------------------------------------------------------

typedef __attribute__((ext_vector_type(16))) __bf16 v16bf;
typedef __attribute__((ext_vector_type(8)))  __bf16 v8bf;
typedef __attribute__((ext_vector_type(8)))  float  v8f;
typedef __attribute__((ext_vector_type(4)))  unsigned int v4u;
typedef __attribute__((ext_vector_type(8)))  int v8i;
typedef __attribute__((ext_vector_type(4)))  int v4i;

#define BB  4
#define SS  2048
#define DD  256
#define HH  8
#define DHH 32
#define SCALE 0.17677669529663687f  // 1/sqrt(DH)

#if defined(__has_builtin)
#if __has_builtin(__builtin_amdgcn_tensor_load_to_lds)
#define USE_TDM 1
#endif
#endif

static __device__ __forceinline__ v8f wmma_bf16(v16bf a, v16bf b, v8f c) {
  // 8 args: (neg_a, A, neg_b, B, c_mod, C, reuse_a, reuse_b)
  return __builtin_amdgcn_wmma_f32_16x16x32_bf16(false, a, false, b,
                                                 (short)0, c, false, false);
}

static __device__ __forceinline__ v16bf cat8(v8bf lo, v8bf hi) {
  return __builtin_shufflevector(lo, hi, 0, 1, 2, 3, 4, 5, 6, 7,
                                 8, 9, 10, 11, 12, 13, 14, 15);
}

#ifdef USE_TDM
// Issue one TDM load of a 32x32 bf16 K tile (rows contiguous, 64B each)
// into LDS at lds_off, padding the LDS destination to 80B rows
// (pad_interval: 16 DWORDs -> code 3; pad_amount: 4 DWORDs -> code 3).
static __device__ __forceinline__ void tdm_load_k_tile(const __bf16* gsrc,
                                                       unsigned int lds_off) {
  const unsigned long long ga = (unsigned long long)(uintptr_t)gsrc;
  v4u g0;
  g0[0] = 1u;                                   // count=1, user descriptor
  g0[1] = lds_off;                              // lds_addr
  g0[2] = (unsigned int)ga;                     // global_addr[31:0]
  g0[3] = (unsigned int)(ga >> 32) | (2u << 30);// global_addr[56:32] | type=2
  v8i g1;
  g1[0] = (3 << 16)   // data_size = 8B units
        | (1 << 20)   // pad_enable
        | (3 << 22)   // pad_interval = 16 DWORDs (64B)
        | (3 << 25);  // pad_amount = 4 DWORDs (16B)
  g1[1] = (int)(8u << 16);    // tensor_dim0 = 8  (64B row, 8B units)
  g1[2] = (int)(32u << 16);   // tensor_dim1 = 32 rows
  g1[3] = (int)(8u << 16);    // tile_dim0 = 8
  g1[4] = 32;                 // tile_dim1 = 32
  g1[5] = 8;                  // tensor_dim0_stride = 8 (contiguous rows)
  g1[6] = 0;
  g1[7] = 0;
  const v4i z4 = {0, 0, 0, 0};            // 2D tensor: groups 2/3 unused
  const v8i z8 = {0, 0, 0, 0, 0, 0, 0, 0};
  // 6-arg form (this toolchain): (g0, g1, g2, g3, g4, cpol)
  __builtin_amdgcn_tensor_load_to_lds(g0, g1, z4, z4, z8, 0);
}
#endif

// ---------------------------------------------------------------------------
// Kernel 0a: f32 -> bf16 activation conversion.
// ---------------------------------------------------------------------------
__global__ void cvt_x_kernel(const float* __restrict__ x,
                             __bf16* __restrict__ xb, int n) {
  int i = blockIdx.x * blockDim.x + threadIdx.x;
  if (i < n) xb[i] = (__bf16)x[i];
}

// ---------------------------------------------------------------------------
// Kernel 0b: transposed bf16 weights WT[w][h][e][d] = W_w[h][d][e]:
// makes the WMMA B-operand one contiguous 16-element bf16 load.
// ---------------------------------------------------------------------------
__global__ void build_wt_kernel(const float* __restrict__ Wq,
                                const float* __restrict__ Wk,
                                const float* __restrict__ Wv,
                                __bf16* __restrict__ wt) {
  int idx = blockIdx.x * blockDim.x + threadIdx.x;
  const int n = 3 * HH * DHH * DD;
  if (idx >= n) return;
  int d = idx % DD;
  int e = (idx / DD) % DHH;
  int h = (idx / (DD * DHH)) % HH;
  int w = idx / (DD * DHH * HH);
  const float* src = (w == 0) ? Wq : (w == 1) ? Wk : Wv;
  wt[idx] = (__bf16)src[((size_t)h * DD + d) * DHH + e];
}

// ---------------------------------------------------------------------------
// Kernel 1: QKV projection. Grid: (S/64, B*H), block 128 (4 waves).
// Q is stored pre-scaled by 1/sqrt(DH); V is stored TRANSPOSED
// (vT[bh][dh][token]) so attention's P*V B-operand is a contiguous load.
// ---------------------------------------------------------------------------
__global__ __launch_bounds__(128) void qkv_proj_kernel(
    const __bf16* __restrict__ xb, const __bf16* __restrict__ wt,
    const float* __restrict__ bq, const float* __restrict__ bk,
    const float* __restrict__ bv,
    __bf16* __restrict__ qg, __bf16* __restrict__ kg, __bf16* __restrict__ vT) {
  const int lane = threadIdx.x & 31;
  const int wave = threadIdx.x >> 5;
  const int hlf  = lane >> 4;
  const int ln   = lane & 15;
  const int bh = blockIdx.y, b = bh / HH, h = bh % HH;
  const int m0 = blockIdx.x * 64 + wave * 16;

  const __bf16* xrow = xb + ((size_t)b * SS + m0 + ln) * DD;
  const __bf16* wq = wt + (size_t)(0 * HH + h) * DHH * DD;
  const __bf16* wk = wt + (size_t)(1 * HH + h) * DHH * DD;
  const __bf16* wv = wt + (size_t)(2 * HH + h) * DHH * DD;

  v8f q0a = {}, q1a = {}, k0a = {}, k1a = {}, v0a = {}, v1a = {};

  for (int k = 0; k < DD; k += 32) {
    // A in this code's contiguous k order (B matches): row M = lane&15, K = 16*half+{0..15}.
    v16bf a = cat8(*(const v8bf*)(xrow + k + 16 * hlf),
                   *(const v8bf*)(xrow + k + 16 * hlf + 8));
    const int kc = k + 16 * hlf;   // 32B-aligned column base
    v16bf bq0 = *(const v16bf*)(wq + (size_t)(ln) * DD + kc);
    v16bf bq1 = *(const v16bf*)(wq + (size_t)(16 + ln) * DD + kc);
    v16bf bk0 = *(const v16bf*)(wk + (size_t)(ln) * DD + kc);
    v16bf bk1 = *(const v16bf*)(wk + (size_t)(16 + ln) * DD + kc);
    v16bf bv0 = *(const v16bf*)(wv + (size_t)(ln) * DD + kc);
    v16bf bv1 = *(const v16bf*)(wv + (size_t)(16 + ln) * DD + kc);
    q0a = wmma_bf16(a, bq0, q0a);
    q1a = wmma_bf16(a, bq1, q1a);
    k0a = wmma_bf16(a, bk0, k0a);
    k1a = wmma_bf16(a, bk1, k1a);
    v0a = wmma_bf16(a, bv0, v0a);
    v1a = wmma_bf16(a, bv1, v1a);
  }

  const float bq0s = bq[h * DHH + ln], bq1s = bq[h * DHH + 16 + ln];
  const float bk0s = bk[h * DHH + ln], bk1s = bk[h * DHH + 16 + ln];
  const float bv0s = bv[h * DHH + ln], bv1s = bv[h * DHH + 16 + ln];

  // C layout: VGPR i holds row M = i + 8*half, col N = lane&15.
  const size_t obase = ((size_t)bh * SS + m0) * DHH;
#pragma unroll
  for (int i = 0; i < 8; ++i) {
    const size_t r = obase + (size_t)(i + 8 * hlf) * DHH;
    qg[r + ln]      = (__bf16)((q0a[i] + bq0s) * SCALE);  // fold softmax scale
    qg[r + 16 + ln] = (__bf16)((q1a[i] + bq1s) * SCALE);
    kg[r + ln]      = (__bf16)(k0a[i] + bk0s);
    kg[r + 16 + ln] = (__bf16)(k1a[i] + bk1s);
  }

  // V transposed: fixed dh = n, 8 consecutive tokens -> one v8bf store.
  v8bf pv0, pv1;
#pragma unroll
  for (int i = 0; i < 8; ++i) {
    pv0[i] = (__bf16)(v0a[i] + bv0s);
    pv1[i] = (__bf16)(v1a[i] + bv1s);
  }
  *(v8bf*)(vT + ((size_t)bh * DHH + ln) * SS + m0 + 8 * hlf)      = pv0;
  *(v8bf*)(vT + ((size_t)bh * DHH + 16 + ln) * SS + m0 + 8 * hlf) = pv1;
}

// ---------------------------------------------------------------------------
// Kernel 2: flash attention. Grid: (S/64, B*H), block 128 (4 waves).
// Wave owns 16 query rows; loops over 64 tiles of 32 keys.
// K tiles: TDM -> double-buffered LDS (shared by all 4 waves).
// V tiles: contiguous v16bf loads from transposed V (L2-resident).
// ---------------------------------------------------------------------------
__global__ __launch_bounds__(128) void attn_kernel(
    const __bf16* __restrict__ qg, const __bf16* __restrict__ kg,
    const __bf16* __restrict__ vT, float* __restrict__ out) {
  // K rows padded to 40 halves (80B): 16B alignment everywhere, and the
  // TDM pad_enable path reproduces exactly this layout.
  __shared__ __attribute__((aligned(16))) __bf16 lK[2][32][40];
  __shared__ __attribute__((aligned(16))) __bf16 lP[4][16][40];

  const int tid = threadIdx.x, lane = tid & 31, wave = tid >> 5;
  const int hlf = lane >> 4, ln = lane & 15;
  const int bh = blockIdx.y, b = bh / HH, h = bh % HH;
  const int m0 = blockIdx.x * 64 + wave * 16;

  // Q A-operand is loop-invariant (already scaled by 1/sqrt(DH)).
  const __bf16* qrow = qg + ((size_t)bh * SS + m0 + ln) * DHH;
  const v16bf qa = cat8(*(const v8bf*)(qrow + 16 * hlf),
                        *(const v8bf*)(qrow + 16 * hlf + 8));

  v8f o0 = {}, o1 = {};
  float rowM[8], rowS[8];
#pragma unroll
  for (int i = 0; i < 8; ++i) { rowM[i] = -3.0e38f; rowS[i] = 0.0f; }

  const __bf16* kbase = kg + (size_t)bh * SS * DHH;
  const __bf16* vtb   = vT + (size_t)bh * DHH * SS;
  const int NIT = SS / 32;

#ifdef USE_TDM
  const unsigned int lkoff[2] = {(unsigned int)(uintptr_t)&lK[0][0][0],
                                 (unsigned int)(uintptr_t)&lK[1][0][0]};
  if (wave == 0) tdm_load_k_tile(kbase, lkoff[0]);  // prologue: tile 0
#endif

  for (int it = 0; it < NIT; ++it) {
    const int ct = it * 32;
    const int buf = it & 1;
#ifdef USE_TDM
    __syncthreads();  // everyone done reading lK[buf^1] (iteration it-1)
    if (wave == 0) {
      if (it + 1 < NIT) {
        tdm_load_k_tile(kbase + (size_t)(ct + 32) * DHH, lkoff[buf ^ 1]);
        __builtin_amdgcn_s_wait_tensorcnt(1);  // tile `it` complete
      } else {
        __builtin_amdgcn_s_wait_tensorcnt(0);
      }
    }
    __syncthreads();  // publish lK[buf] to all waves
#else
    __syncthreads();
    {  // cooperative fallback: 128 threads x 16B = 2KB tile
      const int crow = tid >> 2, cseg = tid & 3;
      *(v8bf*)&lK[buf][crow][cseg * 8] =
          *(const v8bf*)(kbase + (size_t)(ct + crow) * DHH + cseg * 8);
    }
    __syncthreads();
#endif

    // Scores: B = K^T. Lane holds token N = ln (+16 for 2nd tile),
    // dh K-range = 16*half + {0..15} -> contiguous within a K row.
    const v16bf kb0 = cat8(*(const v8bf*)&lK[buf][ln][16 * hlf],
                           *(const v8bf*)&lK[buf][ln][16 * hlf + 8]);
    const v16bf kb1 = cat8(*(const v8bf*)&lK[buf][16 + ln][16 * hlf],
                           *(const v8bf*)&lK[buf][16 + ln][16 * hlf + 8]);
    v8f s0 = {}, s1 = {};
    s0 = wmma_bf16(qa, kb0, s0);
    s1 = wmma_bf16(qa, kb1, s1);

    // P*V B-operand from transposed V: lane holds dh N = ln (+16),
    // token K-range = ct + 16*half + {0..15} -> contiguous, 32B aligned.
    const v16bf vb0 = *(const v16bf*)(vtb + (size_t)ln * SS + ct + 16 * hlf);
    const v16bf vb1 =
        *(const v16bf*)(vtb + (size_t)(16 + ln) * SS + ct + 16 * hlf);
    if (ct + 32 < SS) {  // CDNA5 global_prefetch of next V tile
      __builtin_prefetch(vtb + (size_t)ln * SS + ct + 32 + 16 * hlf, 0, 1);
      __builtin_prefetch(vtb + (size_t)(16 + ln) * SS + ct + 32 + 16 * hlf, 0, 1);
    }

    // Online softmax; shfl_xor masks 1/2/4/8 reduce within the 16-lane
    // C-layout half on wave32.
    float p0v[8], p1v[8];
#pragma unroll
    for (int i = 0; i < 8; ++i) {
      const float a0 = s0[i], a1 = s1[i];  // scale already folded into Q
      float mx = fmaxf(a0, a1);
      mx = fmaxf(mx, __shfl_xor(mx, 1));
      mx = fmaxf(mx, __shfl_xor(mx, 2));
      mx = fmaxf(mx, __shfl_xor(mx, 4));
      mx = fmaxf(mx, __shfl_xor(mx, 8));
      const float mnew = fmaxf(rowM[i], mx);
      const float alpha = __expf(rowM[i] - mnew);
      rowM[i] = mnew;
      const float e0 = __expf(a0 - mnew);
      const float e1 = __expf(a1 - mnew);
      float ps = e0 + e1;
      ps += __shfl_xor(ps, 1);
      ps += __shfl_xor(ps, 2);
      ps += __shfl_xor(ps, 4);
      ps += __shfl_xor(ps, 8);
      rowS[i] = rowS[i] * alpha + ps;
      o0[i] *= alpha;
      o1[i] *= alpha;
      p0v[i] = e0;
      p1v[i] = e1;
    }

    // C-layout -> A-layout transpose of P through this wave's LDS slab.
    // DS ops from one wave are in-order: no barrier needed.
#pragma unroll
    for (int i = 0; i < 8; ++i) {
      const int m = i + 8 * hlf;
      lP[wave][m][ln]      = (__bf16)p0v[i];
      lP[wave][m][16 + ln] = (__bf16)p1v[i];
    }
    const v16bf pa = cat8(*(const v8bf*)&lP[wave][ln][16 * hlf],
                          *(const v8bf*)&lP[wave][ln][16 * hlf + 8]);
    o0 = wmma_bf16(pa, vb0, o0);
    o1 = wmma_bf16(pa, vb1, o1);
  }

  // Epilogue: divide by softmax denominator, scatter heads into [B,S,D].
#pragma unroll
  for (int i = 0; i < 8; ++i) {
    const int m = m0 + i + 8 * hlf;
    const float inv = 1.0f / rowS[i];
    float* orow = out + ((size_t)b * SS + m) * DD + h * DHH;
    orow[ln]      = o0[i] * inv;
    orow[16 + ln] = o1[i] * inv;
  }
}

// ---------------------------------------------------------------------------
// Launch
// ---------------------------------------------------------------------------
extern "C" void kernel_launch(void* const* d_in, const int* in_sizes, int n_in,
                              void* d_out, int out_size, void* d_ws,
                              size_t ws_size, hipStream_t stream) {
  const float* x  = (const float*)d_in[0];
  const float* Wq = (const float*)d_in[1];
  const float* Wk = (const float*)d_in[2];
  const float* Wv = (const float*)d_in[3];
  const float* bq = (const float*)d_in[4];
  const float* bk = (const float*)d_in[5];
  const float* bv = (const float*)d_in[6];

  const size_t nX = (size_t)BB * SS * DD;       // 2,097,152
  const size_t nW = (size_t)3 * HH * DHH * DD;  //   196,608
  const size_t nQ = (size_t)BB * HH * SS * DHH; // 2,097,152
  __bf16* xb   = (__bf16*)d_ws;
  __bf16* wt   = xb + nX;
  __bf16* qbuf = wt + nW;
  __bf16* kbuf = qbuf + nQ;
  __bf16* vtbuf = kbuf + nQ;  // transposed V; total ~16.4 MB workspace

  cvt_x_kernel<<<(int)((nX + 255) / 256), 256, 0, stream>>>(x, xb, (int)nX);
  build_wt_kernel<<<(int)((nW + 255) / 256), 256, 0, stream>>>(Wq, Wk, Wv, wt);

  dim3 grid(SS / 64, BB * HH);  // (32, 32)
  qkv_proj_kernel<<<grid, 128, 0, stream>>>(xb, wt, bq, bk, bv,
                                            qbuf, kbuf, vtbuf);
  attn_kernel<<<grid, 128, 0, stream>>>(qbuf, kbuf, vtbuf, (float*)d_out);
}
